// SemiseparableLayer_71906342469738
// MI455X (gfx1250) — matrix operand order repaired and checked by reading, not compiled, not run on hardware
//
#include <hip/hip_runtime.h>
#include <hip/hip_bf16.h>

// Semiseparable layer for MI455X (gfx1250), wave32 + WMMA bf16.
//
// v3: weights are pre-converted f32->bf16 once per launch into d_ws (8.5 MB);
// the 64-stage scan stages each stage's weights into LDS with
// global_load_async_to_lds_b128 (ASYNCcnt) instead of f32 load + VALU convert
// + ds_store. Falls back to the in-loop f32 conversion path if ws is too
// small or null; without the async builtin the bf16 copy uses plain loads/stores.

#define NB      64
#define SD      128
#define BI      64
#define BO      64
#define BATCH   8192
#define INDIM   4096
#define OUTDIM  4096
#define NTILE   64        // batch columns per workgroup
#define THREADS 256       // 8 wave32 waves

typedef __attribute__((ext_vector_type(16))) __bf16 v16bf;
typedef __attribute__((ext_vector_type(8)))  __bf16 v8bf;
typedef __attribute__((ext_vector_type(8)))  float  v8f;
typedef __attribute__((__vector_size__(4 * sizeof(int)))) int v4i;

// ---- bf16 weight cache layout inside d_ws (element offsets) ----
constexpr size_t N_A  = (size_t)NB * SD * SD;   // A / E
constexpr size_t N_B  = (size_t)NB * SD * BI;   // B / F
constexpr size_t N_C  = (size_t)NB * BO * SD;   // C / G
constexpr size_t N_D  = (size_t)NB * BO * BI;   // D
constexpr size_t OFFA = 0;
constexpr size_t OFFB = OFFA + N_A;
constexpr size_t OFFC = OFFB + N_B;
constexpr size_t OFFD = OFFC + N_C;
constexpr size_t OFFE = OFFD + N_D;
constexpr size_t OFFF = OFFE + N_A;
constexpr size_t OFFG = OFFF + N_B;
constexpr size_t WTOT = OFFG + N_C;             // 4,456,448 bf16 = 8.5 MB

#if defined(__HIP_DEVICE_COMPILE__) && __has_builtin(__builtin_amdgcn_global_load_async_to_lds_b128)
#define HAVE_ASYNC_LDS 1
#else
#define HAVE_ASYNC_LDS 0
#endif

static __device__ inline unsigned short f2bf(float x) {
  __bf16 b = (__bf16)x;
  return __builtin_bit_cast(unsigned short, b);
}
static __device__ inline unsigned long long pack4(float4 v) {
  return  (unsigned long long)f2bf(v.x)
       | ((unsigned long long)f2bf(v.y) << 16)
       | ((unsigned long long)f2bf(v.z) << 32)
       | ((unsigned long long)f2bf(v.w) << 48);
}
static __device__ inline void cvt_store4(__bf16* dst, float4 v) {
  *reinterpret_cast<unsigned long long*>(dst) = pack4(v);
}

// async DMA: 16 bytes global -> LDS per lane, tracked by ASYNCcnt
static __device__ inline void async_copy_b128(const __bf16* g, __bf16* l) {
#if HAVE_ASYNC_LDS
  __builtin_amdgcn_global_load_async_to_lds_b128(
      (__attribute__((address_space(1))) v4i*)g,
      (__attribute__((address_space(3))) v4i*)l, 0, 0);
#else
  *reinterpret_cast<v8bf*>(l) = *reinterpret_cast<const v8bf*>(g);
#endif
}
static __device__ inline void wait_async0() {
#if HAVE_ASYNC_LDS
#if __has_builtin(__builtin_amdgcn_s_wait_asynccnt)
  __builtin_amdgcn_s_wait_asynccnt(0);
#else
  asm volatile("s_wait_asynccnt 0x0" ::: "memory");
#endif
#endif
}

// f32 matrix -> bf16 LDS (same layout), prefetch next stage
static __device__ inline void load_mat_f32(const float* __restrict__ g,
                                           const float* __restrict__ gnext,
                                           __bf16* l, int n, int tid) {
  const float4* g4 = reinterpret_cast<const float4*>(g);
  const int n4 = n >> 2;
  for (int i = tid; i < n4; i += THREADS) {
    cvt_store4(l + i * 4, g4[i]);
    if (gnext && (i & 7) == 0)
      __builtin_prefetch(gnext + i * 4, 0, 1);   // global_prefetch_b8
  }
}

// bf16 matrix (ws cache) -> LDS via async DMA, prefetch next stage
static __device__ inline void load_mat_bf16(const __bf16* __restrict__ g,
                                            const __bf16* __restrict__ gnext,
                                            __bf16* l, int n, int tid) {
  const int n8 = n >> 3;                         // 8 bf16 per b128
  for (int i = tid; i < n8; i += THREADS) {
    async_copy_b128(g + i * 8, l + i * 8);
    if (gnext && (i & 15) == 0)
      __builtin_prefetch(gnext + i * 8, 0, 1);
  }
}

// A-operand fragment (16x32 bf16, row-major weights [m][k], leading dim ld).
// lanes 0-15 hold M=lane, K {k0..k0+7, k0+16..k0+23};
// lanes 16-31 hold M=lane-16, K {k0+8..k0+15, k0+24..k0+31}.
static __device__ inline v16bf frag_a(const __bf16* w, int ld, int row0,
                                      int k0, int lane) {
  const int m  = row0 + (lane & 15);
  const int kk = k0 + ((lane & 16) ? 8 : 0);
  const __bf16* p = w + m * ld + kk;
  v8bf lo = *reinterpret_cast<const v8bf*>(p);
  v8bf hi = *reinterpret_cast<const v8bf*>(p + 16);
  return __builtin_shufflevector(lo, hi, 0, 1, 2, 3, 4, 5, 6, 7,
                                 8, 9, 10, 11, 12, 13, 14, 15);
}

// B-operand fragment (32x16 bf16), matrix stored N-major / K-contiguous.
static __device__ inline v16bf frag_b(const __bf16* s, int ld, int k0,
                                      int n0, int lane) {
  const int n  = n0 + (lane & 15);
  const int kk = k0 + ((lane & 16) ? 8 : 0);
  const __bf16* p = s + n * ld + kk;
  v8bf lo = *reinterpret_cast<const v8bf*>(p);
  v8bf hi = *reinterpret_cast<const v8bf*>(p + 16);
  return __builtin_shufflevector(lo, hi, 0, 1, 2, 3, 4, 5, 6, 7,
                                 8, 9, 10, 11, 12, 13, 14, 15);
}

static __device__ inline v8f wmma_bf16(v16bf a, v16bf b, v8f c) {
  return __builtin_amdgcn_wmma_f32_16x16x32_bf16(
      false, a, false, b, (short)0, c, false, false);
}

// f32 tensor -> bf16 ws cache
__global__ __launch_bounds__(THREADS)
void cvt_kernel(const float* __restrict__ src, unsigned short* __restrict__ dst,
                int n4) {
  const int i = blockIdx.x * THREADS + threadIdx.x;
  if (i >= n4) return;
  float4 v = reinterpret_cast<const float4*>(src)[i];
  reinterpret_cast<unsigned long long*>(dst)[i] = pack4(v);
}

// out[b][j] = bias[j]
__global__ __launch_bounds__(THREADS)
void bias_init_kernel(const float* __restrict__ bias, float* __restrict__ out) {
  const size_t i = (size_t)blockIdx.x * THREADS + threadIdx.x;   // float4 index
  float4 b = reinterpret_cast<const float4*>(bias)[i & (OUTDIM / 4 - 1)];
  reinterpret_cast<float4*>(out)[i] = b;
}

template <bool PRE>
__global__ __launch_bounds__(THREADS)
void semisep_scan_kernel(const float* __restrict__ U,
                         const float* __restrict__ A, const float* __restrict__ B,
                         const float* __restrict__ C, const float* __restrict__ D,
                         const float* __restrict__ E, const float* __restrict__ F,
                         const float* __restrict__ G,
                         const unsigned short* __restrict__ wsw,
                         float* __restrict__ out) {
  __shared__ __attribute__((aligned(32))) __bf16 ldsA[SD * SD];       // 32 KB
  __shared__ __attribute__((aligned(32))) __bf16 ldsB[SD * BI];       // 16 KB
  __shared__ __attribute__((aligned(32))) __bf16 ldsC[BO * SD];       // 16 KB
  __shared__ __attribute__((aligned(32))) __bf16 ldsD[BO * BI];       //  8 KB
  __shared__ __attribute__((aligned(32))) __bf16 ldsS[2][NTILE * SD]; // 32 KB state ping-pong [n][k]
  __shared__ __attribute__((aligned(32))) __bf16 ldsU[NTILE * BI];    //  8 KB u tile [n][k]
  __shared__ __attribute__((aligned(16))) float  ldsY[NTILE * BO];    // 16 KB y staging [n][m]

  const int  tid  = threadIdx.x;
  const int  lane = tid & 31;
  const int  wv   = tid >> 5;                 // 0..7
  const int  b0   = blockIdx.x * NTILE;       // batch tile origin
  const bool caus = (blockIdx.y == 0);

  for (int i = tid; i < NTILE * SD; i += THREADS) ldsS[0][i] = (__bf16)0.0f;

  int cur = 0;
  #pragma unroll 1
  for (int t = 0; t < NB; ++t) {
    const int  k    = caus ? t : (NB - 1 - t);
    const int  kn   = caus ? (k + 1) : (k - 1);
    const bool last = (t == NB - 1);

    __syncthreads();   // previous stage fully consumed LDS weights / Y / state

    if constexpr (PRE) {
      const __bf16* wb = reinterpret_cast<const __bf16*>(wsw);
      const __bf16* wA = wb + (caus ? OFFA : OFFE) + (size_t)k * SD * SD;
      const __bf16* wB = wb + (caus ? OFFB : OFFF) + (size_t)k * SD * BI;
      const __bf16* wC = wb + (caus ? OFFC : OFFG) + (size_t)k * BO * SD;
      const __bf16* wAN = last ? nullptr : wb + (caus ? OFFA : OFFE) + (size_t)kn * SD * SD;
      const __bf16* wBN = last ? nullptr : wb + (caus ? OFFB : OFFF) + (size_t)kn * SD * BI;
      const __bf16* wCN = last ? nullptr : wb + (caus ? OFFC : OFFG) + (size_t)kn * BO * SD;
      load_mat_bf16(wA, wAN, ldsA, SD * SD, tid);
      load_mat_bf16(wB, wBN, ldsB, SD * BI, tid);
      load_mat_bf16(wC, wCN, ldsC, BO * SD, tid);
      if (caus) {
        const __bf16* wD  = wb + OFFD + (size_t)k * BO * BI;
        const __bf16* wDN = last ? nullptr : wb + OFFD + (size_t)kn * BO * BI;
        load_mat_bf16(wD, wDN, ldsD, BO * BI, tid);
      }
    } else {
      const float* Wa = (caus ? A : E) + (size_t)k * SD * SD;
      const float* Wb = (caus ? B : F) + (size_t)k * SD * BI;
      const float* Wc = (caus ? C : G) + (size_t)k * BO * SD;
      const float* WaN = last ? nullptr : (caus ? A : E) + (size_t)kn * SD * SD;
      const float* WbN = last ? nullptr : (caus ? B : F) + (size_t)kn * SD * BI;
      const float* WcN = last ? nullptr : (caus ? C : G) + (size_t)kn * BO * SD;
      load_mat_f32(Wa, WaN, ldsA, SD * SD, tid);
      load_mat_f32(Wb, WbN, ldsB, SD * BI, tid);
      load_mat_f32(Wc, WcN, ldsC, BO * SD, tid);
      if (caus) {
        const float* Wd  = D + (size_t)k * BO * BI;
        const float* WdN = last ? nullptr : D + (size_t)kn * BO * BI;
        load_mat_f32(Wd, WdN, ldsD, BO * BI, tid);
      }
    }

    {   // u_k tile: ldsU[n][i] = bf16(U[(b0+n)*INDIM + k*BI + i])
      const float* up = U + (size_t)b0 * INDIM + (size_t)k * BI;
      for (int i = tid; i < NTILE * (BI / 4); i += THREADS) {
        const int row = i >> 4, c4 = i & 15;
        float4 v = *reinterpret_cast<const float4*>(up + (size_t)row * INDIM + c4 * 4);
        cvt_store4(ldsU + row * BI + c4 * 4, v);
      }
    }
    if constexpr (PRE) wait_async0();
    __syncthreads();

    const __bf16* S = ldsS[cur];
    const v8f z = {};

    // ---- state update: wave wv owns rows [16wv,16wv+16) x all 64 batch cols
    v8f sa0 = z, sa1 = z, sa2 = z, sa3 = z;
    #pragma unroll
    for (int ks = 0; ks < 4; ++ks) {          // A (or E) @ state, K = 128
      v16bf a = frag_a(ldsA, SD, 16 * wv, ks * 32, lane);
      sa0 = wmma_bf16(a, frag_b(S, SD, ks * 32, 0,  lane), sa0);
      sa1 = wmma_bf16(a, frag_b(S, SD, ks * 32, 16, lane), sa1);
      sa2 = wmma_bf16(a, frag_b(S, SD, ks * 32, 32, lane), sa2);
      sa3 = wmma_bf16(a, frag_b(S, SD, ks * 32, 48, lane), sa3);
    }
    #pragma unroll
    for (int ks = 0; ks < 2; ++ks) {          // + B (or F) @ u, K = 64
      v16bf a = frag_a(ldsB, BI, 16 * wv, ks * 32, lane);
      sa0 = wmma_bf16(a, frag_b(ldsU, BI, ks * 32, 0,  lane), sa0);
      sa1 = wmma_bf16(a, frag_b(ldsU, BI, ks * 32, 16, lane), sa1);
      sa2 = wmma_bf16(a, frag_b(ldsU, BI, ks * 32, 32, lane), sa2);
      sa3 = wmma_bf16(a, frag_b(ldsU, BI, ks * 32, 48, lane), sa3);
    }

    // ---- y: wave wv owns M-tile (wv>>1), two N tiles at n0, n0+16
    const int mt = wv >> 1;
    const int n0 = (wv & 1) * 32;
    v8f y0 = z, y1 = z;
    #pragma unroll
    for (int ks = 0; ks < 4; ++ks) {          // C (or G) @ state
      v16bf a = frag_a(ldsC, SD, 16 * mt, ks * 32, lane);
      y0 = wmma_bf16(a, frag_b(S, SD, ks * 32, n0,      lane), y0);
      y1 = wmma_bf16(a, frag_b(S, SD, ks * 32, n0 + 16, lane), y1);
    }
    if (caus) {
      #pragma unroll
      for (int ks = 0; ks < 2; ++ks) {        // + D @ u
        v16bf a = frag_a(ldsD, BI, 16 * mt, ks * 32, lane);
        y0 = wmma_bf16(a, frag_b(ldsU, BI, ks * 32, n0,      lane), y0);
        y1 = wmma_bf16(a, frag_b(ldsU, BI, ks * 32, n0 + 16, lane), y1);
      }
    }

    // stash y tiles to LDS (f32), [n][m]
    {
      const int nn = lane & 15, mh = (lane & 16) ? 8 : 0;
      #pragma unroll
      for (int r = 0; r < 8; ++r) {
        const int m = 16 * mt + r + mh;
        ldsY[(n0      + nn) * BO + m] = y0[r];
        ldsY[(n0 + 16 + nn) * BO + m] = y1[r];
      }
    }
    __syncthreads();   // all reads of ldsS[cur] done; ldsY complete

    // write new state (bf16) into the other ping-pong buffer
    {
      const int nn = lane & 15, mh = (lane & 16) ? 8 : 0;
      __bf16* Sn = ldsS[cur ^ 1];
      #pragma unroll
      for (int r = 0; r < 8; ++r) {
        const int m = 16 * wv + r + mh;
        Sn[(0  + nn) * SD + m] = (__bf16)sa0[r];
        Sn[(16 + nn) * SD + m] = (__bf16)sa1[r];
        Sn[(32 + nn) * SD + m] = (__bf16)sa2[r];
        Sn[(48 + nn) * SD + m] = (__bf16)sa3[r];
      }
    }

    // flush y tile: coalesced atomic f32 adds into d_out
    {
      float* op = out + (size_t)b0 * OUTDIM + (size_t)k * BO;
      for (int i = tid; i < NTILE * BO; i += THREADS) {
        const int n = i >> 6, m = i & 63;
        unsafeAtomicAdd(op + (size_t)n * OUTDIM + m, ldsY[n * BO + m]);
      }
    }

    cur ^= 1;
  }
}

extern "C" void kernel_launch(void* const* d_in, const int* in_sizes, int n_in,
                              void* d_out, int out_size, void* d_ws, size_t ws_size,
                              hipStream_t stream) {
  const float* U    = (const float*)d_in[0];
  const float* A    = (const float*)d_in[1];
  const float* B    = (const float*)d_in[2];
  const float* C    = (const float*)d_in[3];
  const float* D    = (const float*)d_in[4];
  const float* E    = (const float*)d_in[5];
  const float* F    = (const float*)d_in[6];
  const float* G    = (const float*)d_in[7];
  const float* bias = (const float*)d_in[8];
  float* out = (float*)d_out;
  (void)in_sizes; (void)n_in; (void)out_size;

  const bool pre = (d_ws != nullptr) && (ws_size >= WTOT * sizeof(unsigned short));
  unsigned short* wsw = (unsigned short*)d_ws;

  if (pre) {  // weights f32 -> bf16 cache in d_ws (runs every call; deterministic)
    auto cvt = [&](const float* s, size_t off, size_t n) {
      const int n4 = (int)(n >> 2);
      cvt_kernel<<<(n4 + THREADS - 1) / THREADS, THREADS, 0, stream>>>(s, wsw + off, n4);
    };
    cvt(A, OFFA, N_A); cvt(B, OFFB, N_B); cvt(C, OFFC, N_C); cvt(D, OFFD, N_D);
    cvt(E, OFFE, N_A); cvt(F, OFFF, N_B); cvt(G, OFFG, N_C);
  }

  // out = broadcast(bias)
  const int init_blocks = (BATCH * (OUTDIM / 4)) / THREADS;   // 32768
  bias_init_kernel<<<init_blocks, THREADS, 0, stream>>>(bias, out);

  // causal (y=0) and anticausal (y=1) scans in parallel
  dim3 grid(BATCH / NTILE, 2);
  if (pre)
    semisep_scan_kernel<true><<<grid, THREADS, 0, stream>>>(U, A, B, C, D, E, F, G, wsw, out);
  else
    semisep_scan_kernel<false><<<grid, THREADS, 0, stream>>>(U, A, B, C, D, E, F, G, wsw, out);
}
